// NodeFeatures_18047452578374
// MI455X (gfx1250) — matrix operand order repaired and checked by reading, not compiled, not run on hardware
//
#include <hip/hip_runtime.h>
#include <hip/hip_bf16.h>

typedef __attribute__((ext_vector_type(16))) __bf16 v16bf;
typedef __attribute__((ext_vector_type(8)))  float  v8f;

#define D_FEAT 128
#define H_FEAT 256

union Frag16 { v16bf v; uint32_t u[8]; };

// ---------------------------------------------------------------------------
// Prep kernels
// ---------------------------------------------------------------------------
__global__ void cvt_bf16_kernel(const float* __restrict__ src,
                                __bf16* __restrict__ dst, long n) {
  long i = (long)blockIdx.x * blockDim.x + threadIdx.x;
  if (i < n) dst[i] = (__bf16)src[i];
}

// W is K x Ncol row-major; write Wt as Ncol x K row-major (bf16)
__global__ void transpose_bf16_kernel(const float* __restrict__ W,
                                      __bf16* __restrict__ Wt, int K, int Ncol) {
  int i = blockIdx.x * blockDim.x + threadIdx.x;
  if (i < K * Ncol) {
    int k = i / Ncol, n = i % Ncol;           // coalesced read of W
    Wt[(long)n * K + k] = (__bf16)W[i];
  }
}

__global__ void init_accum_kernel(float* __restrict__ denom,
                                  float* __restrict__ agg, long n) {
  long i = (long)blockIdx.x * blockDim.x + threadIdx.x;
  if (i < n) { denom[i] = 1e-7f; agg[i] = 0.0f; }
}

// tiny global-feature FCNN: g3 = relu(g @ W3a + b3a) @ W3b + b3b
__global__ void g3_kernel(const float* __restrict__ g,
                          const float* __restrict__ W3a, const float* __restrict__ b3a,
                          const float* __restrict__ W3b, const float* __restrict__ b3b,
                          float* __restrict__ g3) {
  __shared__ float t[H_FEAT];
  int h = threadIdx.x;                         // 256 threads
  float acc = b3a[h];
  for (int d = 0; d < D_FEAT; ++d) acc += g[d] * W3a[d * H_FEAT + h];
  t[h] = fmaxf(acc, 0.0f);
  __syncthreads();
  if (h < D_FEAT) {
    float a2 = b3b[h];
    for (int k = 0; k < H_FEAT; ++k) a2 += t[k] * W3b[k * D_FEAT + h];
    g3[h] = a2;
  }
}

// ---------------------------------------------------------------------------
// WMMA GEMM: C[M x Ncol] = act(A[M x K] @ B[K x Ncol] + bias)
// A: bf16 row-major.  Bt: bf16, B transposed (Ncol x K row-major).
// Block: 256 threads (8 waves). Block tile 128(M) x 64(N).
// Wave w owns rows w*16..w*16+15 and all 64 cols (4 WMMA accumulator tiles).
// ---------------------------------------------------------------------------
#define BM 128
#define BN 64
#define BK 32
#define LDK (BK + 8)   // +8 bf16 (16B) pad: keeps uint4 alignment, kills bank conflicts

template <bool RELU, bool STOREBF>
__global__ void gemm_wmma_kernel(const __bf16* __restrict__ A,
                                 const __bf16* __restrict__ Bt,
                                 const float* __restrict__ bias,
                                 float* __restrict__ Cf, __bf16* __restrict__ Cb,
                                 int M, int Ncol, int K) {
  __shared__ __align__(16) __bf16 lA[BM][LDK];   // 10 KB
  __shared__ __align__(16) __bf16 lB[BN][LDK];   // 5 KB

  const int m0   = blockIdx.x * BM;
  const int n0   = blockIdx.y * BN;
  const int t    = threadIdx.x;
  const int wave = t >> 5;
  const int lane = t & 31;
  const int half = lane >> 4;   // K-group selector per ISA 16-bit layouts
  const int l16  = lane & 15;

  v8f acc[4];
  #pragma unroll
  for (int i = 0; i < 4; ++i)
    #pragma unroll
    for (int r = 0; r < 8; ++r) acc[i][r] = 0.0f;

  for (int k0 = 0; k0 < K; k0 += BK) {
    // --- stage A tile (128x32): thread t -> row t/2, 16-elem half t%2
    {
      int row = t >> 1;
      int hk  = (t & 1) * 16;
      uint4 v0 = make_uint4(0u, 0u, 0u, 0u), v1 = v0;
      if (m0 + row < M) {
        const __bf16* s = A + (long)(m0 + row) * K + k0 + hk;
        v0 = *(const uint4*)(s);
        v1 = *(const uint4*)(s + 8);
      }
      *(uint4*)(&lA[row][hk])     = v0;
      *(uint4*)(&lA[row][hk + 8]) = v1;
    }
    // --- stage Bt tile (64x32): thread t -> row t/4, 8-elem chunk t%4
    {
      int n  = t >> 2;
      int ck = (t & 3) * 8;
      *(uint4*)(&lB[n][ck]) =
          *(const uint4*)(Bt + (long)(n0 + n) * K + k0 + ck);
    }
    __syncthreads();

    // --- A fragment (ISA 16-bit A 16x32 layout):
    // lane half=0: VGPR p<4 -> K=2p..2p+1, p>=4 -> K=16+2(p-4)..; half=1: +8
    Frag16 fa;
    const uint32_t* arow = (const uint32_t*)(&lA[wave * 16 + l16][0]);
    #pragma unroll
    for (int p = 0; p < 8; ++p) {
      int kk = (p < 4) ? (half * 8 + 2 * p) : (16 + half * 8 + 2 * (p - 4));
      fa.u[p] = arow[kk >> 1];
    }

    #pragma unroll
    for (int tile = 0; tile < 4; ++tile) {
      // --- B fragment: col = tile*16 + l16, VGPR p holds same K as A fragment
      Frag16 fb;
      const uint32_t* brow = (const uint32_t*)(&lB[tile * 16 + l16][0]);
      #pragma unroll
      for (int p = 0; p < 8; ++p) fb.u[p] = brow[((p < 4) ? (half * 8 + 2 * p) : (16 + half * 8 + 2 * (p - 4))) >> 1];

      acc[tile] = __builtin_amdgcn_wmma_f32_16x16x32_bf16(
          false, fa.v, false, fb.v, (short)0, acc[tile], false, false);
    }
    __syncthreads();
  }

  // --- epilogue (ISA f32 C layout: lanes<16 -> M=r, lanes>=16 -> M=8+r)
  #pragma unroll
  for (int tile = 0; tile < 4; ++tile) {
    int col  = n0 + tile * 16 + l16;
    float bv = bias[col];
    #pragma unroll
    for (int r = 0; r < 8; ++r) {
      int row = m0 + wave * 16 + half * 8 + r;
      if (row < M) {
        float v = acc[tile][r] + bv;
        if (RELU) v = fmaxf(v, 0.0f);
        if (STOREBF) Cb[(long)row * Ncol + col] = (__bf16)v;
        else         Cf[(long)row * Ncol + col] = v;
      }
    }
  }
}

// ---------------------------------------------------------------------------
// Fused edge pass: s = sigmoid(ef); denom[src]+=s; denom[dst]+=s;
// agg[src]+=s*h2[dst]; agg[dst]+=s*h2[src].  128 threads per edge, 2 edges/block.
// ---------------------------------------------------------------------------
__global__ void edge_kernel(const float* __restrict__ ef, const int* __restrict__ idx,
                            const float* __restrict__ h2, float* __restrict__ agg,
                            float* __restrict__ denom, int E) {
  int t = threadIdx.x;
  int e = blockIdx.x * 2 + (t >> 7);
  int d = t & (D_FEAT - 1);
  if (e >= E) return;
  int src = idx[e];
  int dst = idx[E + e];
  __builtin_prefetch(ef + (long)(e + 16) * D_FEAT + d, 0, 0);  // stream ahead
  float s = 1.0f / (1.0f + __expf(-ef[(long)e * D_FEAT + d]));
  float hs = h2[(long)src * D_FEAT + d];
  float hd = h2[(long)dst * D_FEAT + d];
  unsafeAtomicAdd(&denom[(long)src * D_FEAT + d], s);
  unsafeAtomicAdd(&denom[(long)dst * D_FEAT + d], s);
  unsafeAtomicAdd(&agg[(long)src * D_FEAT + d], s * hd);
  unsafeAtomicAdd(&agg[(long)dst * D_FEAT + d], s * hs);
}

// ---------------------------------------------------------------------------
// Finalize: inter = h1 + agg/denom + g3; InstanceNorm over D=128 (one wave32
// per row, 4 feats/lane, shfl_xor reduction); out = x + relu(normed)
// ---------------------------------------------------------------------------
__global__ void finalize_kernel(const float* __restrict__ x, const float* __restrict__ h1,
                                const float* __restrict__ agg, const float* __restrict__ denom,
                                const float* __restrict__ g3, float* __restrict__ out, int Nn) {
  int wave = threadIdx.x >> 5;
  int lane = threadIdx.x & 31;
  int row  = blockIdx.x * 8 + wave;
  if (row >= Nn) return;
  long base = (long)row * D_FEAT;
  float v[4], s = 0.0f, sq = 0.0f;
  #pragma unroll
  for (int j = 0; j < 4; ++j) {
    int d = lane + 32 * j;
    float val = h1[base + d] + agg[base + d] / denom[base + d] + g3[d];
    v[j] = val; s += val; sq += val * val;
  }
  #pragma unroll
  for (int m = 16; m >= 1; m >>= 1) {
    s  += __shfl_xor(s,  m, 32);
    sq += __shfl_xor(sq, m, 32);
  }
  float mean = s * (1.0f / D_FEAT);
  float var  = sq * (1.0f / D_FEAT) - mean * mean;
  float inv  = rsqrtf(var + 1e-5f);
  #pragma unroll
  for (int j = 0; j < 4; ++j) {
    int d = lane + 32 * j;
    float nv = (v[j] - mean) * inv;
    out[base + d] = x[base + d] + fmaxf(nv, 0.0f);
  }
}

// ---------------------------------------------------------------------------
// Launch
// ---------------------------------------------------------------------------
extern "C" void kernel_launch(void* const* d_in, const int* in_sizes, int n_in,
                              void* d_out, int out_size, void* d_ws, size_t ws_size,
                              hipStream_t stream) {
  const float* x   = (const float*)d_in[0];
  const int*   idx = (const int*)  d_in[1];
  const float* ef  = (const float*)d_in[2];
  const float* gf  = (const float*)d_in[3];
  const float* W1a = (const float*)d_in[4];  const float* b1a = (const float*)d_in[5];
  const float* W1b = (const float*)d_in[6];  const float* b1b = (const float*)d_in[7];
  const float* W2a = (const float*)d_in[8];  const float* b2a = (const float*)d_in[9];
  const float* W2b = (const float*)d_in[10]; const float* b2b = (const float*)d_in[11];
  const float* W3a = (const float*)d_in[12]; const float* b3a = (const float*)d_in[13];
  const float* W3b = (const float*)d_in[14]; const float* b3b = (const float*)d_in[15];
  float* out = (float*)d_out;

  const int Nn = in_sizes[0] / D_FEAT;
  const int E  = in_sizes[2] / D_FEAT;
  const long ND = (long)Nn * D_FEAT;
  const long NH = (long)Nn * H_FEAT;
  const long DH = (long)D_FEAT * H_FEAT;

  // workspace carve-up (256B aligned slices)
  char* p = (char*)d_ws;
  auto take = [&](long bytes) { char* r = p; p += (bytes + 255) & ~255L; return r; };
  __bf16* xbf  = (__bf16*)take(ND * 2);
  __bf16* w1at = (__bf16*)take(DH * 2);
  __bf16* w1bt = (__bf16*)take(DH * 2);
  __bf16* w2at = (__bf16*)take(DH * 2);
  __bf16* w2bt = (__bf16*)take(DH * 2);
  __bf16* ybf  = (__bf16*)take(NH * 2);
  float*  h1   = (float*) take(ND * 4);
  float*  h2   = (float*) take(ND * 4);
  float*  den  = (float*) take(ND * 4);
  float*  agg  = (float*) take(ND * 4);
  float*  g3   = (float*) take(D_FEAT * 4);

  // 1) conversions / transposes
  cvt_bf16_kernel<<<(int)((ND + 255) / 256), 256, 0, stream>>>(x, xbf, ND);
  int wblk = (int)((DH + 255) / 256);
  transpose_bf16_kernel<<<wblk, 256, 0, stream>>>(W1a, w1at, D_FEAT, H_FEAT);
  transpose_bf16_kernel<<<wblk, 256, 0, stream>>>(W1b, w1bt, H_FEAT, D_FEAT);
  transpose_bf16_kernel<<<wblk, 256, 0, stream>>>(W2a, w2at, D_FEAT, H_FEAT);
  transpose_bf16_kernel<<<wblk, 256, 0, stream>>>(W2b, w2bt, H_FEAT, D_FEAT);

  // 2) accumulator init + tiny global FCNN
  init_accum_kernel<<<(int)((ND + 255) / 256), 256, 0, stream>>>(den, agg, ND);
  g3_kernel<<<1, H_FEAT, 0, stream>>>(gf, W3a, b3a, W3b, b3b, g3);

  // 3) node FCNNs on the WMMA pipe (bf16 in, f32 accumulate)
  dim3 gH((Nn + BM - 1) / BM, H_FEAT / BN);
  dim3 gD((Nn + BM - 1) / BM, D_FEAT / BN);
  gemm_wmma_kernel<true,  true ><<<gH, 256, 0, stream>>>(xbf, w1at, b1a, nullptr, ybf, Nn, H_FEAT, D_FEAT);
  gemm_wmma_kernel<false, false><<<gD, 256, 0, stream>>>(ybf, w1bt, b1b, h1, nullptr, Nn, D_FEAT, H_FEAT);
  gemm_wmma_kernel<true,  true ><<<gH, 256, 0, stream>>>(xbf, w2at, b2a, nullptr, ybf, Nn, H_FEAT, D_FEAT);
  gemm_wmma_kernel<false, false><<<gD, 256, 0, stream>>>(ybf, w2bt, b2b, h2, nullptr, Nn, D_FEAT, H_FEAT);

  // 4) bandwidth-bound edge scatter (L2-resident f32 atomics)
  edge_kernel<<<(E + 1) / 2, 256, 0, stream>>>(ef, idx, h2, agg, den, E);

  // 5) normalize + residual
  finalize_kernel<<<(Nn + 7) / 8, 256, 0, stream>>>(x, h1, agg, den, g3, out, Nn);
}
